// L2SelfAttention_5566277615865
// MI455X (gfx1250) — compile-verified, matrix operand order checked by reading, not run on hardware
//
#include <hip/hip_runtime.h>
#include <hip/hip_bf16.h>

// Problem constants (from reference: B=2, N=2048, D=1024, H=16)
#define B_  2
#define N_  2048
#define D_  1024
#define H_  16
#define HD_ 64
#define M_  (B_ * N_)   // 4096 token rows

typedef __attribute__((ext_vector_type(16))) __bf16 v16bf;
typedef __attribute__((ext_vector_type(8)))  float  v8f;
typedef __attribute__((ext_vector_type(4)))  unsigned int v4u;
typedef __attribute__((ext_vector_type(8)))  int v8i;
typedef __attribute__((ext_vector_type(4)))  int v4i;

// Padded LDS row strides (elements) for conflict-free 16-lane x 16B reads
// against the 64-bank LDS.
#define GEMM_STRIDE 40   // 32 data + 8 pad  (80B rows)
#define KT_STRIDE   72   // 64 data + 8 pad  (144B rows)
#define VT_STRIDE   40   // 32 data + 8 pad  (80B rows)

#if defined(__has_builtin)
#if __has_builtin(__builtin_amdgcn_tensor_load_to_lds)
#define HAVE_TDM 1
#else
#define HAVE_TDM 0
#endif
#else
#define HAVE_TDM 0
#endif

// ---------------------------------------------------------------------------
// TDM: DMA a 2D tile (tile1 rows x tile0 elements of bf16, global row stride
// stride0 elements) into LDS with post-row padding so rows land on a padded
// stride. padCode: LDS pad interval code ((1<<code) 8-byte units between
// 16B pads): 3 -> every 64B (row len 32 elems), 4 -> every 128B (64 elems).
// ---------------------------------------------------------------------------
__device__ __forceinline__ void tdm_load_2d(const void* gsrc, void* lds,
                                            unsigned dim0, unsigned dim1,
                                            unsigned tile0, unsigned tile1,
                                            unsigned stride0, unsigned padCode) {
#if HAVE_TDM
  const unsigned long long ga = (unsigned long long)(uintptr_t)gsrc;
  const unsigned lda = (unsigned)(uintptr_t)lds;   // low 32 bits = LDS offset
  v4u g0;
  g0[0] = 1u;                                      // count=1, user descriptor
  g0[1] = lda;                                     // lds_addr (bytes)
  g0[2] = (unsigned)(ga & 0xFFFFFFFFu);            // global_addr[31:0]
  g0[3] = (unsigned)(ga >> 32) | (2u << 30);       // global_addr[56:32] | type=2
  v8i g1;
  g1[0] = (int)((1u << 16)        // data_size = 2 bytes
              | (1u << 20)        // pad_enable
              | (padCode << 22)   // pad_interval
              | (3u << 25));      // pad_amount: 4 DWORDs = 16B
  g1[1] = (int)(dim0 << 16);                    // tensor_dim0 low16 @ [63:48]
  g1[2] = (int)((dim0 >> 16) | (dim1 << 16));   // dim0 hi | tensor_dim1 lo
  g1[3] = (int)((dim1 >> 16) | (tile0 << 16));  // dim1 hi | tile_dim0
  g1[4] = (int)tile1;                           // tile_dim1 | tile_dim2=0
  g1[5] = (int)stride0;                         // tensor_dim0_stride low32
  g1[6] = 0;                                    // stride0 hi | stride1 lo
  g1[7] = 0;
  v4i z = {0, 0, 0, 0};
#if __clang_major__ >= 23
  v8i z8 = {0, 0, 0, 0, 0, 0, 0, 0};
  __builtin_amdgcn_tensor_load_to_lds(g0, g1, z, z, z8, 0);
#else
  __builtin_amdgcn_tensor_load_to_lds(g0, g1, z, z, 0);
#endif
#else
  (void)gsrc; (void)lds; (void)dim0; (void)dim1;
  (void)tile0; (void)tile1; (void)stride0; (void)padCode;
#endif
}

// ---------------------------------------------------------------------------
// Fragment loaders. 16x32 bf16 A/B fragment, 16 elems/lane:
// lane l<16 covers row l, K = {k..k+7, k+16..k+23}; lane l>=16 covers row l-16,
// K = {k+8..k+15, k+24..k+31}. Contiguous 16B runs.
// ---------------------------------------------------------------------------
static __device__ __forceinline__ v16bf lds_frag(const __bf16* tile, int stride,
                                                 int k0, int lane) {
  const int half = lane >> 4, l = lane & 15;
  const __bf16* p = tile + l * stride + k0 + half * 8;
  v16bf f;
#pragma unroll
  for (int i = 0; i < 8; ++i) f[i] = p[i];
#pragma unroll
  for (int i = 0; i < 8; ++i) f[8 + i] = p[16 + i];
  return f;
}

static __device__ __forceinline__ v16bf glb_frag(const __bf16* tile, int ld,
                                                 int k0, int lane) {
  const int half = lane >> 4, l = lane & 15;
  const __bf16* p = tile + (size_t)l * ld + k0 + half * 8;
  v16bf f;
#pragma unroll
  for (int i = 0; i < 8; ++i) f[i] = p[i];
#pragma unroll
  for (int i = 0; i < 8; ++i) f[8 + i] = p[16 + i];
  return f;
}

static __device__ __forceinline__ v8f wmma_bf16(v16bf a, v16bf b, v8f c) {
  return __builtin_amdgcn_wmma_f32_16x16x32_bf16(false, a, false, b,
                                                 (short)0, c, false, false);
}

// Generic cooperative fallback copy (rows x 8*segs 16B chunks), 256 threads.
template <int ROWS, int SEGS, int DSTRIDE>
__device__ __forceinline__ void coop_copy(const __bf16* g, int ldg,
                                          __bf16* lds, int t) {
  for (int id = t; id < ROWS * SEGS; id += 256) {
    const int row = id / SEGS, seg = id % SEGS;
    const __bf16* src = g + (size_t)row * ldg + seg * 8;
    __bf16* dst = lds + row * DSTRIDE + seg * 8;
#pragma unroll
    for (int u = 0; u < 8; ++u) dst[u] = src[u];
  }
}

// ---------------------------------------------------------------------------
// f32 -> bf16 conversion
// ---------------------------------------------------------------------------
__global__ void cvt_bf16_kernel(const float* __restrict__ src,
                                __bf16* __restrict__ dst, int n) {
  int i = blockIdx.x * blockDim.x + threadIdx.x;
  if (i < n) dst[i] = (__bf16)src[i];
}

// ---------------------------------------------------------------------------
// GEMM: C[M_, D_] = A[M_, D_](bf16) @ W[D_, D_]^T(bf16) + bias
// Block 256 / 8 waves; block tile 128x128; wave tile 32x64 (2x4 wmma tiles).
// K-step 32 staged through double-buffered padded LDS via TDM.
// mode 0/1: bf16 Q/K as (B,H,N,hd) + fused row sum-of-squares -> ss.
// mode 2:   bf16 V transposed per head (B,H,hd,N).
// mode 3:   f32 + bias (final output projection).
// ---------------------------------------------------------------------------
__global__ __launch_bounds__(256) void gemm_proj_kernel(
    const __bf16* __restrict__ A, const __bf16* __restrict__ W,
    const float* __restrict__ bias, __bf16* __restrict__ outb,
    float* __restrict__ outf, float* __restrict__ ss, int mode) {
  __shared__ __align__(128) __bf16 Abuf[2][128 * GEMM_STRIDE];
  __shared__ __align__(128) __bf16 Bbuf[2][128 * GEMM_STRIDE];

  const int t = threadIdx.x;
  const int lane = t & 31, wave = t >> 5;
  const int rowBlock = blockIdx.y * 128;
  const int colBlock = blockIdx.x * 128;
  const int rowW = (wave >> 1) * 32;
  const int colW = (wave & 1) * 64;

  auto stage = [&](const __bf16* g, __bf16* lds) {
#if HAVE_TDM
    if (wave == 0) tdm_load_2d(g, lds, D_, M_, 32, 128, D_, 3u);
#else
    coop_copy<128, 4, GEMM_STRIDE>(g, D_, lds, t);
#endif
  };

  stage(A + (size_t)rowBlock * D_, Abuf[0]);
  stage(W + (size_t)colBlock * D_, Bbuf[0]);

  v8f c[2][4] = {};
  const int steps = D_ / 32;
  for (int i = 0; i < steps; ++i) {
    const int buf = i & 1;
#if HAVE_TDM
    if (wave == 0) __builtin_amdgcn_s_wait_tensorcnt((short)0);
#endif
    __syncthreads();
    if (i + 1 < steps) {
      const int k0 = (i + 1) * 32;
      stage(A + (size_t)rowBlock * D_ + k0, Abuf[buf ^ 1]);
      stage(W + (size_t)colBlock * D_ + k0, Bbuf[buf ^ 1]);
    }
    v16bf bfrag[4];
#pragma unroll
    for (int j = 0; j < 4; ++j)
      bfrag[j] = lds_frag(Bbuf[buf] + (colW + j * 16) * GEMM_STRIDE,
                          GEMM_STRIDE, 0, lane);
#pragma unroll
    for (int ii = 0; ii < 2; ++ii) {
      v16bf afrag = lds_frag(Abuf[buf] + (rowW + ii * 16) * GEMM_STRIDE,
                             GEMM_STRIDE, 0, lane);
#pragma unroll
      for (int j = 0; j < 4; ++j)
        c[ii][j] = wmma_bf16(afrag, bfrag[j], c[ii][j]);
    }
  }

  const int half = lane >> 4, l15 = lane & 15;
  float bv[4];
#pragma unroll
  for (int j = 0; j < 4; ++j) bv[j] = bias[colBlock + colW + j * 16 + l15];
#pragma unroll
  for (int ii = 0; ii < 2; ++ii) {
#pragma unroll
    for (int e = 0; e < 8; ++e) {
      const int row = rowBlock + rowW + ii * 16 + e + 8 * half;
      const int b = row / N_, n = row % N_;
      float s = 0.f;
#pragma unroll
      for (int j = 0; j < 4; ++j) {
        const int col = colBlock + colW + j * 16 + l15;
        const float v = c[ii][j][e] + bv[j];
        if (mode == 3) {
          outf[(size_t)row * D_ + col] = v;
        } else {
          const int h = col >> 6, cc = col & 63;
          if (mode == 2)
            outb[(((size_t)(b * H_ + h)) * HD_ + cc) * N_ + n] = (__bf16)v;
          else
            outb[(((size_t)(b * H_ + h)) * N_ + n) * HD_ + cc] = (__bf16)v;
        }
        s += v * v;
      }
      if (mode <= 1) {   // fused ||q||^2 / ||k||^2 (wave spans one full head)
        const int h = (colBlock + colW) >> 6;
#pragma unroll
        for (int m = 1; m <= 8; m <<= 1) s += __shfl_xor(s, m, 32);
        if (l15 == 0) ss[((size_t)(b * H_ + h)) * N_ + n] = s;
      }
    }
  }
}

// ---------------------------------------------------------------------------
// Fused L2-distance attention, S^T formulation.
// grid (N/128, B*H), block 256 = 8 waves; wave owns 16 queries.
// Per 32-key step (K/V tiles TDM-staged in LDS, shared by all 8 waves):
//   S^T = K * Q^T  (2x2 wmma; C-fragment: lane = query, elems = keys)
//   score = -sqrt(q2 + k2 - 2 S^T): per-lane online softmax, one shfl_xor(16)
//   per reduction. The S^T C-layout IS the P A-layout -> P fragment is built
//   in registers, no LDS transpose. O += P V (4 wmma, V pre-transposed).
// ---------------------------------------------------------------------------
__global__ __launch_bounds__(256) void attn_kernel(
    const __bf16* __restrict__ Q, const __bf16* __restrict__ Kb,
    const __bf16* __restrict__ Vt, const float* __restrict__ q2,
    const float* __restrict__ k2, __bf16* __restrict__ O) {
  __shared__ __align__(128) __bf16 Ktile[2][32 * KT_STRIDE];  // 32 keys x 64 hd
  __shared__ __align__(128) __bf16 Vtile[2][64 * VT_STRIDE];  // 64 hd x 32 keys

  const int t = threadIdx.x;
  const int lane = t & 31, wave = t >> 5;
  const int half = lane >> 4, l15 = lane & 15;
  const int bh = blockIdx.y;
  const int q0 = blockIdx.x * 128 + wave * 16;

  const __bf16* Qh  = Q  + (size_t)bh * N_ * HD_;
  const __bf16* Kh  = Kb + (size_t)bh * N_ * HD_;
  const __bf16* Vh  = Vt + (size_t)bh * HD_ * N_;   // (hd, N)
  const float*  q2h = q2 + (size_t)bh * N_;
  const float*  k2h = k2 + (size_t)bh * N_;

  auto stageK = [&](int j, __bf16* lds) {
#if HAVE_TDM
    if (wave == 0)
      tdm_load_2d(Kh + (size_t)j * HD_, lds, HD_, N_, HD_, 32, HD_, 4u);
#else
    coop_copy<32, 8, KT_STRIDE>(Kh + (size_t)j * HD_, HD_, lds, t);
#endif
  };
  auto stageV = [&](int j, __bf16* lds) {
#if HAVE_TDM
    if (wave == 0)
      tdm_load_2d(Vh + j, lds, N_, HD_, 32, HD_, N_, 3u);
#else
    coop_copy<64, 4, VT_STRIDE>(Vh + j, N_, lds, t);
#endif
  };

  // Q fragments: B operand of S^T = K Q^T (same memory layout as A use).
  v16bf qf[2];
  qf[0] = glb_frag(Qh + (size_t)q0 * HD_, HD_, 0,  lane);
  qf[1] = glb_frag(Qh + (size_t)q0 * HD_, HD_, 32, lane);
  const float q2v = q2h[q0 + l15];   // this lane's query norm

  float mcur = -INFINITY, lcur = 0.f;
  v8f acc[4] = {};

  stageK(0, Ktile[0]);
  stageV(0, Vtile[0]);

  for (int j = 0; j < N_; j += 32) {
    const int buf = (j >> 5) & 1;
#if HAVE_TDM
    if (wave == 0) __builtin_amdgcn_s_wait_tensorcnt((short)0);
#endif
    __syncthreads();
    if (j + 32 < N_) {
      stageK(j + 32, Ktile[buf ^ 1]);
      stageV(j + 32, Vtile[buf ^ 1]);
    }

    // S^T tiles: rows = keys, cols = queries.
    v8f st0 = {}, st1 = {};
#pragma unroll
    for (int kk = 0; kk < 2; ++kk) {
      v16bf ka = lds_frag(Ktile[buf],                  KT_STRIDE, kk * 32, lane);
      v16bf kb = lds_frag(Ktile[buf] + 16 * KT_STRIDE, KT_STRIDE, kk * 32, lane);
      st0 = wmma_bf16(ka, qf[kk], st0);
      st1 = wmma_bf16(kb, qf[kk], st1);
    }

    // Per-lane scores: 16 keys (e+8*half and 16+e+8*half) for query l15.
    float tv[16];
#pragma unroll
    for (int e = 0; e < 8; ++e) {
      const float k2a = k2h[j + 8 * half + e];
      const float k2b = k2h[j + 16 + 8 * half + e];
      tv[e]     = -sqrtf(fmaxf(q2v + k2a - 2.f * st0[e], 1e-12f));
      tv[8 + e] = -sqrtf(fmaxf(q2v + k2b - 2.f * st1[e], 1e-12f));
    }
    float mx = tv[0];
#pragma unroll
    for (int e = 1; e < 16; ++e) mx = fmaxf(mx, tv[e]);
    mx = fmaxf(mx, __shfl_xor(mx, 16, 32));
    const float mnew = fmaxf(mcur, mx);
    const float alpha = __expf(mcur - mnew);
    mcur = mnew;

    float rs = 0.f;
    v16bf pf;                 // P A-fragment == S^T C-fragment layout: free!
#pragma unroll
    for (int e = 0; e < 16; ++e) {
      const float p = __expf(tv[e] - mnew);
      rs += p;
      pf[e] = (__bf16)p;
    }
    rs += __shfl_xor(rs, 16, 32);
    lcur = lcur * alpha + rs;

    // Broadcast alpha to accumulator row owners and rescale O.
    float arow[8];
#pragma unroll
    for (int e = 0; e < 8; ++e) arow[e] = __shfl(alpha, e + 8 * half, 32);
#pragma unroll
    for (int cc = 0; cc < 4; ++cc)
#pragma unroll
      for (int e = 0; e < 8; ++e) acc[cc][e] *= arow[e];

    // O += P(16x32) * V(32x64) from the shared LDS V tile.
#pragma unroll
    for (int cc = 0; cc < 4; ++cc) {
      v16bf vf = lds_frag(Vtile[buf] + (cc * 16) * VT_STRIDE, VT_STRIDE, 0, lane);
      acc[cc] = wmma_bf16(pf, vf, acc[cc]);
    }
  }

  // Epilogue: normalize, write O into (B, N, H*hd) bf16 for the out-proj.
  float linv[8];
#pragma unroll
  for (int e = 0; e < 8; ++e)
    linv[e] = __builtin_amdgcn_rcpf(__shfl(lcur, e + 8 * half, 32));
  const int bb = bh / H_, h = bh % H_;
#pragma unroll
  for (int cc = 0; cc < 4; ++cc) {
#pragma unroll
    for (int e = 0; e < 8; ++e) {
      const int row = q0 + e + 8 * half;
      const int col = h * HD_ + cc * 16 + l15;
      O[((size_t)(bb * N_ + row)) * D_ + col] = (__bf16)(acc[cc][e] * linv[e]);
    }
  }
}

// ---------------------------------------------------------------------------
extern "C" void kernel_launch(void* const* d_in, const int* in_sizes, int n_in,
                              void* d_out, int out_size, void* d_ws, size_t ws_size,
                              hipStream_t stream) {
  (void)in_sizes; (void)n_in; (void)out_size; (void)ws_size;
  const float* x  = (const float*)d_in[0];
  const float* wq = (const float*)d_in[1];
  const float* bq = (const float*)d_in[2];
  const float* wk = (const float*)d_in[3];
  const float* bk = (const float*)d_in[4];
  const float* wv = (const float*)d_in[5];
  const float* bv = (const float*)d_in[6];
  const float* wo = (const float*)d_in[7];
  const float* bo = (const float*)d_in[8];

  char* ws = (char*)d_ws;
  const size_t SZ_ACT = (size_t)M_ * D_ * sizeof(__bf16);  // 8 MiB
  const size_t SZ_W   = (size_t)D_ * D_ * sizeof(__bf16);  // 2 MiB
  const size_t SZ_SS  = (size_t)B_ * H_ * N_ * sizeof(float);
  __bf16* xb  = (__bf16*)ws; ws += SZ_ACT;
  __bf16* wqb = (__bf16*)ws; ws += SZ_W;
  __bf16* wkb = (__bf16*)ws; ws += SZ_W;
  __bf16* wvb = (__bf16*)ws; ws += SZ_W;
  __bf16* wob = (__bf16*)ws; ws += SZ_W;
  __bf16* Qb  = (__bf16*)ws; ws += SZ_ACT;
  __bf16* Kb  = (__bf16*)ws; ws += SZ_ACT;
  __bf16* Vtb = (__bf16*)ws; ws += SZ_ACT;
  __bf16* Ob  = (__bf16*)ws; ws += SZ_ACT;
  float*  q2  = (float*)ws;  ws += SZ_SS;
  float*  k2  = (float*)ws;  ws += SZ_SS;

  const int nAct = M_ * D_, nW = D_ * D_;
  cvt_bf16_kernel<<<nAct / 256, 256, 0, stream>>>(x,  xb,  nAct);
  cvt_bf16_kernel<<<nW   / 256, 256, 0, stream>>>(wq, wqb, nW);
  cvt_bf16_kernel<<<nW   / 256, 256, 0, stream>>>(wk, wkb, nW);
  cvt_bf16_kernel<<<nW   / 256, 256, 0, stream>>>(wv, wvb, nW);
  cvt_bf16_kernel<<<nW   / 256, 256, 0, stream>>>(wo, wob, nW);

  dim3 gg(D_ / 128, M_ / 128);   // (8, 32)
  gemm_proj_kernel<<<gg, 256, 0, stream>>>(xb, wqb, bq, Qb,  nullptr, q2, 0);
  gemm_proj_kernel<<<gg, 256, 0, stream>>>(xb, wkb, bk, Kb,  nullptr, k2, 1);
  gemm_proj_kernel<<<gg, 256, 0, stream>>>(xb, wvb, bv, Vtb, nullptr, nullptr, 2);

  dim3 ga(N_ / 128, B_ * H_);    // (16, 32)
  attn_kernel<<<ga, 256, 0, stream>>>(Qb, Kb, Vtb, q2, k2, Ob);

  gemm_proj_kernel<<<gg, 256, 0, stream>>>(Ob, wob, bo, nullptr, (float*)d_out,
                                           nullptr, 3);
}
